// TSPRGCNActionNet_53360673686184
// MI455X (gfx1250) — matrix operand order repaired and checked by reading, not compiled, not run on hardware
//
#include <hip/hip_runtime.h>
#include <hip/hip_bf16.h>

// ---------------------------------------------------------------------------
// TSP RGCN action net for MI455X (gfx1250, wave32).
// Memory-bound (~2.5 GB traffic vs ~60 GFLOP @ 23.3 TB/s): all 128-wide GEMMs
// go through v_wmma_f32_16x16x32_f16 (f32 accum). Weights are pre-packed once
// to f16 transposed chunk layout; each GEMM block stages a 32 KB panel into
// LDS via the Tensor Data Mover (pad_enable inserts the 16B/row bank padding),
// B fragments are two ds_load_b128, A fragments contiguous global_load_b128.
// MLP activations kept in f16 to halve HBM traffic.
// ---------------------------------------------------------------------------

typedef __attribute__((ext_vector_type(16))) _Float16 v16h;
typedef __attribute__((ext_vector_type(8)))  _Float16 v8h;
typedef __attribute__((ext_vector_type(8)))  float    v8f;
typedef __attribute__((ext_vector_type(4)))  float    v4f;
typedef __attribute__((ext_vector_type(4)))  unsigned int v4u;
typedef __attribute__((ext_vector_type(8)))  int      v8i;
typedef __attribute__((ext_vector_type(4)))  int      v4i;

constexpr int NB = 24;     // batch
constexpr int NV = 100;    // nodes
constexpr int NH = 128;    // hidden
constexpr int NP = NV * (NV - 1) / 2;  // 4950 pairs per batch
constexpr int LDT = 136;   // LDS stride (halfs) of transposed panel: 272B, 16B-aligned

#if defined(__has_builtin)
#if __has_builtin(__builtin_amdgcn_tensor_load_to_lds) && __has_builtin(__builtin_amdgcn_s_wait_tensorcnt)
#define USE_TDM 1
#endif
#endif

__device__ __forceinline__ v16h cat16(v8h a, v8h b) {
  return __builtin_shufflevector(a, b, 0, 1, 2, 3, 4, 5, 6, 7, 8, 9, 10, 11, 12, 13, 14, 15);
}
__device__ __forceinline__ v8h cvt8(v4f a, v4f b) {
  v8h r;
  r[0] = (_Float16)a[0]; r[1] = (_Float16)a[1]; r[2] = (_Float16)a[2]; r[3] = (_Float16)a[3];
  r[4] = (_Float16)b[0]; r[5] = (_Float16)b[1]; r[6] = (_Float16)b[2]; r[7] = (_Float16)b[3];
  return r;
}

// ---------------------------------------------------------------------------
// One-time weight pack: W[K x 128] f32 row-major -> Wp chunk layout, f16:
//   Wp[(kc*128 + n)*128 + kr]  (kc = k/128, kr = k%128)   (B-fragment order)
// ---------------------------------------------------------------------------
__global__ void k_pack_w(const float* __restrict__ W, _Float16* __restrict__ Wp, int K) {
  const int idx = blockIdx.x * blockDim.x + threadIdx.x;  // over K*128
  if (idx >= K * 128) return;
  const int k = idx >> 7, n = idx & 127;  // consecutive threads sweep n: coalesced reads
  const int kc = k >> 7, kr = k & 127;
  Wp[((long)kc * 128 + n) * 128 + kr] = (_Float16)W[idx];
}

// ---------------------------------------------------------------------------
// Generic WMMA GEMM: O[M x 128] = act(A[M x K] @ W + bias), W pre-packed f16.
// One wave computes a 16-row tile across all 8 column tiles; block = 8 waves
// = 128 rows. K must be a multiple of 128 (here 128 or 640).
// ---------------------------------------------------------------------------
template <typename TA, typename TO, bool RELU>
__global__ void gemm_n128(const TA* __restrict__ A, int M, int lda,
                          const _Float16* __restrict__ Wp, const float* __restrict__ bias,
                          int K, TO* __restrict__ O, int ldc) {
  __shared__ __attribute__((aligned(16))) _Float16 lwT[128 * LDT];  // 34.8 KB panel
  const int lane = threadIdx.x & 31;
  const int wave = threadIdx.x >> 5;            // 0..7
  const int r0   = blockIdx.x * 128 + wave * 16;

  v8f acc[8];
  const v8f vzero = {0.f, 0.f, 0.f, 0.f, 0.f, 0.f, 0.f, 0.f};
#pragma unroll
  for (int nt = 0; nt < 8; ++nt) acc[nt] = vzero;

  const int rowA = r0 + (lane & 15);
  const int khiA = (lane >> 4) << 3;            // 0 or 8  (A frag K offset, halfs)
  const int khiB = (lane >> 4) << 3;            // 0 or 8  (B frag K offset, halfs)

  for (int kc = 0; kc < K; kc += 128) {
    const _Float16* src = Wp + (size_t)(kc >> 7) * 128 * 128;  // 32 KB chunk
#ifdef USE_TDM
    // TDM: 2D tile 128 rows x 256B, pad 4 dwords every 64 dwords -> LDS stride
    // 272B = LDT halfs. Wave 0 issues; EXEC-independent, one issue per wave.
    if (wave == 0) {
      const unsigned long long ga = (unsigned long long)(size_t)src;
      const unsigned ldsOff = (unsigned)(size_t)(void*)lwT;  // LDS aperture: low 32 bits = LDS byte offset
      v4u g0;
      g0[0] = 1u;                                   // count=1 (valid descriptor)
      g0[1] = ldsOff;                               // lds_addr
      g0[2] = (unsigned)ga;                         // global_addr[31:0]
      g0[3] = (unsigned)((ga >> 32) & 0x1FFFFFFull) // global_addr[56:32]
              | (2u << 30);                         // type=2 ("image")
      v8i g1;
      g1[0] = (int)((1u << 16)    // data_size = 2 bytes
                    | (1u << 20)  // pad_enable
                    | (5u << 22)  // pad_interval: every 64 dwords (256B)
                    | (3u << 25)); // pad_amount: 4 dwords (16B)
      g1[1] = (int)(128u << 16);  // tensor_dim0 = 128 (dim0 low bits at [63:48])
      g1[2] = (int)(128u << 16);  // tensor_dim1 = 128 (low 16 at [95:80])
      g1[3] = (int)(128u << 16);  // tile_dim0 = 128 ([127:112])
      g1[4] = 128;                // tile_dim1 = 128 ([143:128])
      g1[5] = 128;                // tensor_dim0_stride = 128 ([207:160] low)
      g1[6] = 0;
      g1[7] = 0;
      v4i gz4;
      gz4[0] = 0; gz4[1] = 0; gz4[2] = 0; gz4[3] = 0;
      v8i gz8;
      gz8[0] = 0; gz8[1] = 0; gz8[2] = 0; gz8[3] = 0;
      gz8[4] = 0; gz8[5] = 0; gz8[6] = 0; gz8[7] = 0;
      __builtin_amdgcn_tensor_load_to_lds(g0, g1, gz4, gz4, gz8, 0);
      __builtin_amdgcn_s_wait_tensorcnt(0);
    }
    __syncthreads();
#else
    // fallback: straight b128 copies with padded rows
    for (int i = threadIdx.x; i < 128 * 16; i += blockDim.x) {
      const int n = i >> 4, q = i & 15;
      *(v4u*)((char*)lwT + n * (LDT * 2) + q * 16) =
          ((const v4u*)src)[(size_t)n * 16 + q];
    }
    __syncthreads();
#endif

#pragma unroll
    for (int ks = 0; ks < 4; ++ks) {
      const int kb = kc + ks * 32;
      v16h af;
      if (rowA < M) {
        const TA* ap = A + (long)rowA * lda + kb + khiA;
        if constexpr (sizeof(TA) == 2) {
          const v8h g0 = *(const v8h*)(ap);
          const v8h g1 = *(const v8h*)(ap + 16);
          af = cat16(g0, g1);
        } else {
          const v4f f0 = *(const v4f*)(ap);
          const v4f f1 = *(const v4f*)(ap + 4);
          const v4f f2 = *(const v4f*)(ap + 16);
          const v4f f3 = *(const v4f*)(ap + 20);
          af = cat16(cvt8(f0, f1), cvt8(f2, f3));
        }
      } else {
#pragma unroll
        for (int j = 0; j < 16; ++j) af[j] = (_Float16)0.f;
      }
#pragma unroll
      for (int nt = 0; nt < 8; ++nt) {
        const int n = nt * 16 + (lane & 15);
        const _Float16* bp = &lwT[n * LDT + ks * 32 + khiB];
        const v8h b0 = *(const v8h*)(bp);
        const v8h b1 = *(const v8h*)(bp + 16);
        acc[nt] = __builtin_amdgcn_wmma_f32_16x16x32_f16(
            false, af, false, cat16(b0, b1), (short)0, acc[nt], false, false);
      }
    }
    __syncthreads();
  }

  // epilogue: D layout — row = r + 8*(lane>>4), col = lane&15 per 16-col tile
  const int colb  = lane & 15;
  const int rbase = (lane >> 4) << 3;
#pragma unroll
  for (int nt = 0; nt < 8; ++nt) {
    const int col = nt * 16 + colb;
    const float bb = bias[col];
#pragma unroll
    for (int r = 0; r < 8; ++r) {
      const int row = r0 + rbase + r;
      if (row < M) {
        float v = acc[nt][r] + bb;
        if (RELU) v = v > 0.f ? v : 0.f;
        O[(long)row * ldc + col] = (TO)v;
      }
    }
  }
}

// ---------------------------------------------------------------------------
// Embeddings
// ---------------------------------------------------------------------------
__global__ void k_node_embed(const float* __restrict__ coord,
                             const float* __restrict__ w, const float* __restrict__ b,
                             float* __restrict__ x) {
  const int row = blockIdx.x, h = threadIdx.x;
  const float c0 = coord[row * 2 + 0], c1 = coord[row * 2 + 1];
  x[(long)row * NH + h] = c0 * w[h] + c1 * w[NH + h] + b[h];
}

__global__ void k_edge_embed(const float* __restrict__ vals, const float* __restrict__ tour,
                             const float* __restrict__ best,
                             const float* __restrict__ ew, const float* __restrict__ eb,
                             const float* __restrict__ cw, const float* __restrict__ cb,
                             float* __restrict__ e) {
  const long idx = blockIdx.x;
  const int h = threadIdx.x;
  float out;
  if (h < 64) {
    out = vals[idx] * ew[h] + eb[h];
  } else {
    const int j = h - 64;
    out = tour[idx] * cw[j] + best[idx] * cw[64 + j] + cb[j];
  }
  e[idx * NH + h] = out;
}

__global__ void k_zero(float* p, int n) {
  const int i = blockIdx.x * blockDim.x + threadIdx.x;
  if (i < n) p[i] = 0.f;
}

// ---------------------------------------------------------------------------
// Gated aggregation: block owns one (b,u) row of e_tmp; adds broadcast Ax/Bx,
// computes sigmoid gates, reduces over v for agg num/den, accumulates BN stats.
// ---------------------------------------------------------------------------
__global__ void k_edge_gate_agg(float* __restrict__ etmp,
                                const float* __restrict__ Ax, const float* __restrict__ Bx,
                                const float* __restrict__ Vx,
                                float* __restrict__ aggn, float* __restrict__ aggd,
                                float* __restrict__ bnsum, float* __restrict__ bnsq) {
  const int bu = blockIdx.x;
  const int b = bu / NV;
  const int h = threadIdx.x;
  const float ax = Ax[(long)bu * NH + h];
  float an = 0.f, ad = 0.f, se = 0.f, sq = 0.f;
  const long base = (long)bu * NV * NH;
  for (int v = 0; v < NV; ++v) {
    float val = etmp[base + (long)v * NH + h] + ax + Bx[((long)b * NV + v) * NH + h];
    etmp[base + (long)v * NH + h] = val;
    const float g = 1.f / (1.f + __expf(-val));
    an += g * Vx[((long)b * NV + v) * NH + h];
    ad += g;
    se += val;
    sq += val * val;
  }
  aggn[(long)bu * NH + h] = an;
  aggd[(long)bu * NH + h] = ad;
  atomicAdd(&bnsum[h], se);
  atomicAdd(&bnsq[h], sq);
}

__global__ void k_x_update(const float* __restrict__ Ux, const float* __restrict__ aggn,
                           const float* __restrict__ aggd, float* __restrict__ xt,
                           float* __restrict__ bns, float* __restrict__ bnq) {
  const int row = blockIdx.x, h = threadIdx.x;
  const float v = Ux[(long)row * NH + h] +
                  aggn[(long)row * NH + h] / (aggd[(long)row * NH + h] + 1e-20f);
  xt[(long)row * NH + h] = v;
  atomicAdd(&bns[h], v);
  atomicAdd(&bnq[h], v * v);
}

__global__ void k_bn_apply_x(float* __restrict__ x, const float* __restrict__ xt,
                             const float* __restrict__ bns, const float* __restrict__ bnq,
                             const float* __restrict__ g, const float* __restrict__ bb) {
  const int row = blockIdx.x, h = threadIdx.x;
  const float cnt = (float)(NB * NV);
  const float m = bns[h] / cnt;
  const float var = bnq[h] / cnt - m * m;
  float v = (xt[(long)row * NH + h] - m) * rsqrtf(var + 1e-5f) * g[h] + bb[h];
  if (v < 0.f) v = 0.f;
  x[(long)row * NH + h] += v;
}

__global__ void k_bn_apply_e(float* __restrict__ e, const float* __restrict__ et,
                             const float* __restrict__ bns, const float* __restrict__ bnq,
                             const float* __restrict__ g, const float* __restrict__ bb,
                             long n) {
  const float cnt = (float)((long)NB * NV * NV);
  for (long i = (long)blockIdx.x * blockDim.x + threadIdx.x; i < n;
       i += (long)gridDim.x * blockDim.x) {
    const int h = (int)(i & (NH - 1));
    const float m = bns[h] / cnt;
    const float var = bnq[h] / cnt - m * m;
    float v = (et[i] - m) * rsqrtf(var + 1e-5f) * g[h] + bb[h];
    if (v < 0.f) v = 0.f;
    e[i] += v;
  }
}

// ---------------------------------------------------------------------------
// Tour edge list (row-major nonzero over upper triangle; exactly NV per batch)
// ---------------------------------------------------------------------------
__global__ void k_tour_edges(const float* __restrict__ tour, int* __restrict__ tiu,
                             int* __restrict__ tiv) {
  if (threadIdx.x != 0) return;
  const int b = blockIdx.x;
  const long base = (long)b * NV * NV;
  int cnt = 0;
  for (int u = 0; u < NV && cnt < NV; ++u)
    for (int v = u + 1; v < NV; ++v)
      if (tour[base + (long)u * NV + v] > 0.f) {
        tiu[b * NV + cnt] = u;
        tiv[b * NV + cnt] = v;
        ++cnt;
      }
}

// ---------------------------------------------------------------------------
// Quad gather: builds feat[:,0:512] (4 gathered e rows, f16), 2-opt cost, and
// acl hidden layer h1 = relu(cost*w+b).
// ---------------------------------------------------------------------------
__global__ void k_quad_gather(const float* __restrict__ e, const float* __restrict__ vals,
                              const unsigned char* __restrict__ xdir,
                              const int* __restrict__ tiu, const int* __restrict__ tiv,
                              const float* __restrict__ aw, const float* __restrict__ ab,
                              _Float16* __restrict__ feat, _Float16* __restrict__ hb) {
  const int b = blockIdx.x / NP;
  const int p = blockIdx.x % NP;
  const int h = threadIdx.x;
  int r = 0, rem = p, span = NV - 1;
  while (rem >= span) { rem -= span; ++r; --span; }
  const int c = r + 1 + rem;
  int us = tiu[b * NV + r], vs = tiv[b * NV + r];
  int xs = tiu[b * NV + c], ys = tiv[b * NV + c];
  const long bvv = (long)b * NV * NV;
  if (!xdir[bvv + (long)us * NV + vs]) { int t = us; us = vs; vs = t; }
  if (!xdir[bvv + (long)xs * NV + ys]) { int t = xs; xs = ys; ys = t; }
  const long eb = bvv * NH;
  const long row = (long)blockIdx.x;
  _Float16* f = feat + row * 640;
  f[h]       = (_Float16)e[eb + ((long)us * NV + vs) * NH + h];
  f[128 + h] = (_Float16)e[eb + ((long)xs * NV + ys) * NH + h];
  f[256 + h] = (_Float16)e[eb + ((long)us * NV + xs) * NH + h];
  f[384 + h] = (_Float16)e[eb + ((long)vs * NV + ys) * NH + h];
  const float cost = vals[bvv + (long)us * NV + xs] + vals[bvv + (long)vs * NV + ys] -
                     vals[bvv + (long)us * NV + vs] - vals[bvv + (long)xs * NV + ys];
  float h1 = cost * aw[h] + ab[h];
  if (h1 < 0.f) h1 = 0.f;
  hb[row * NH + h] = (_Float16)h1;
}

// final act layer: 128 -> 1
__global__ void k_rowdot(const _Float16* __restrict__ t, const float* __restrict__ w,
                         const float* __restrict__ b, float* __restrict__ logits) {
  __shared__ float s[NH];
  const int row = blockIdx.x, h = threadIdx.x;
  s[h] = (float)t[(long)row * NH + h] * w[h];
  __syncthreads();
  for (int k = 64; k > 0; k >>= 1) {
    if (h < k) s[h] += s[h + k];
    __syncthreads();
  }
  if (h == 0) logits[row] = s[0] + b[0];
}

// per-batch argmax + log_softmax at argmax + edge recovery
__global__ void k_select(const float* __restrict__ logits, const int* __restrict__ tiu,
                         const int* __restrict__ tiv, float* __restrict__ out) {
  __shared__ float sm[256];
  __shared__ int si[256];
  const int b = blockIdx.x, t = threadIdx.x;
  float m = -3.4e38f;
  int mi = 0;
  for (int p = t; p < NP; p += 256) {
    const float l = logits[(long)b * NP + p];
    if (l > m) { m = l; mi = p; }
  }
  sm[t] = m; si[t] = mi;
  __syncthreads();
  for (int s = 128; s > 0; s >>= 1) {
    if (t < s) {
      if (sm[t + s] > sm[t] || (sm[t + s] == sm[t] && si[t + s] < si[t])) {
        sm[t] = sm[t + s]; si[t] = si[t + s];
      }
    }
    __syncthreads();
  }
  const float mx = sm[0];
  const int a = si[0];
  __syncthreads();
  float sum = 0.f;
  for (int p = t; p < NP; p += 256) sum += __expf(logits[(long)b * NP + p] - mx);
  sm[t] = sum;
  __syncthreads();
  for (int s = 128; s > 0; s >>= 1) {
    if (t < s) sm[t] += sm[t + s];
    __syncthreads();
  }
  if (t == 0) {
    const float pi = -logf(sm[0]);  // logits[a] - mx == 0
    int r = 0, rem = a, span = NV - 1;
    while (rem >= span) { rem -= span; ++r; --span; }
    const int c = r + 1 + rem;
    out[b * 6 + 0] = (float)b;
    out[b * 6 + 1] = (float)tiu[b * NV + r];
    out[b * 6 + 2] = (float)tiv[b * NV + r];
    out[b * 6 + 3] = (float)b;
    out[b * 6 + 4] = (float)tiu[b * NV + c];
    out[b * 6 + 5] = (float)tiv[b * NV + c];
    out[NB * 6 + b] = pi;
    out[NB * 6 + NB + b] = (float)a;
  }
}

// ---------------------------------------------------------------------------
// Host orchestration
// ---------------------------------------------------------------------------
extern "C" void kernel_launch(void* const* d_in, const int* in_sizes, int n_in,
                              void* d_out, int out_size, void* d_ws, size_t ws_size,
                              hipStream_t stream) {
  (void)in_sizes; (void)n_in; (void)out_size; (void)ws_size;
  // input map (setup_inputs insertion order, params flattened recursively w,b)
  const float* vals   = (const float*)d_in[1];
  const float* coords = (const float*)d_in[2];
  const float* tour   = (const float*)d_in[3];
  const float* best   = (const float*)d_in[4];
  const unsigned char* xdir = (const unsigned char*)d_in[5];
  auto F = [&](int i) { return (const float*)d_in[i]; };

  // workspace bump allocator
  char* ws = (char*)d_ws;
  size_t off = 0;
  auto alloc = [&](size_t bytes) -> void* {
    off = (off + 255) & ~(size_t)255;
    void* p = ws + off;
    off += bytes;
    return p;
  };
  const size_t eBytes = (size_t)NB * NV * NV * NH * sizeof(float);   // 122.9 MB
  const size_t xBytes = (size_t)NB * NV * NH * sizeof(float);
  float* e    = (float*)alloc(eBytes);
  float* etmp = (float*)alloc(eBytes);
  float* x    = (float*)alloc(xBytes);
  float* Axb  = (float*)alloc(xBytes);
  float* Bxb  = (float*)alloc(xBytes);
  float* Vxb  = (float*)alloc(xBytes);
  float* Uxb  = (float*)alloc(xBytes);
  float* aggn = (float*)alloc(xBytes);
  float* aggd = (float*)alloc(xBytes);
  float* xtmp = (float*)alloc(xBytes);
  float* stats = (float*)alloc(4 * NH * sizeof(float));  // e_sum,e_sq,x_sum,x_sq
  int* tiu = (int*)alloc((size_t)NB * NV * sizeof(int));
  int* tiv = (int*)alloc((size_t)NB * NV * sizeof(int));
  const long Mrows = (long)NB * NP;  // 118800
  _Float16* feat = (_Float16*)alloc((size_t)Mrows * 640 * sizeof(_Float16));
  _Float16* hb   = (_Float16*)alloc((size_t)Mrows * NH * sizeof(_Float16));
  _Float16* s1   = (_Float16*)alloc((size_t)Mrows * NH * sizeof(_Float16));
  _Float16* s2   = (_Float16*)alloc((size_t)Mrows * NH * sizeof(_Float16));
  float* logits  = (float*)alloc((size_t)Mrows * sizeof(float));

  // pre-pack all weight matrices to f16 transposed chunk layout (one-time)
  auto packW = [&](const float* Wsrc, int K) -> _Float16* {
    _Float16* p = (_Float16*)alloc((size_t)K * 128 * sizeof(_Float16));
    const int n = K * 128;
    k_pack_w<<<(n + 255) / 256, 256, 0, stream>>>(Wsrc, p, K);
    return p;
  };
  _Float16 *WpA[3], *WpB[3], *WpC[3], *WpU[3], *WpV[3];
  for (int l = 0; l < 3; ++l) {
    const int p0 = 12 + 14 * l;
    WpA[l] = packW(F(p0 + 0), NH);
    WpB[l] = packW(F(p0 + 2), NH);
    WpC[l] = packW(F(p0 + 4), NH);
    WpU[l] = packW(F(p0 + 6), NH);
    WpV[l] = packW(F(p0 + 8), NH);
  }
  _Float16* Wacl1 = packW(F(56), NH);
  _Float16* Wpre0 = packW(F(58), 640);
  _Float16* Wpre1 = packW(F(60), NH);
  _Float16* Wact[4] = {packW(F(62), NH), packW(F(64), NH), packW(F(66), NH), packW(F(68), NH)};

  // --- embeds ---
  k_node_embed<<<NB * NV, NH, 0, stream>>>(coords, F(6), F(7), x);
  k_edge_embed<<<NB * NV * NV, NH, 0, stream>>>(vals, tour, best, F(8), F(9), F(10), F(11), e);

  // --- 3 gated GCN layers ---
  const int gx  = (NB * NV + 127) / 128;          // 19 blocks for 2400 rows
  const int gE  = (NB * NV * NV + 127) / 128;     // 1875 blocks for 240000 rows
  for (int l = 0; l < 3; ++l) {
    const int p0 = 12 + 14 * l;  // A.w,A.b,B.w,B.b,C.w,C.b,U.w,U.b,V.w,V.b,bnxg,bnxb,bneg,bneb
    gemm_n128<float, float, false><<<gx, 256, 0, stream>>>(x, NB * NV, NH, WpA[l], F(p0 + 1), NH, Axb, NH);
    gemm_n128<float, float, false><<<gx, 256, 0, stream>>>(x, NB * NV, NH, WpB[l], F(p0 + 3), NH, Bxb, NH);
    gemm_n128<float, float, false><<<gx, 256, 0, stream>>>(x, NB * NV, NH, WpV[l], F(p0 + 9), NH, Vxb, NH);
    gemm_n128<float, float, false><<<gx, 256, 0, stream>>>(x, NB * NV, NH, WpU[l], F(p0 + 7), NH, Uxb, NH);
    gemm_n128<float, float, false><<<gE, 256, 0, stream>>>(e, NB * NV * NV, NH, WpC[l], F(p0 + 5), NH, etmp, NH);
    k_zero<<<2, 256, 0, stream>>>(stats, 4 * NH);
    k_edge_gate_agg<<<NB * NV, NH, 0, stream>>>(etmp, Axb, Bxb, Vxb, aggn, aggd, stats, stats + NH);
    k_x_update<<<NB * NV, NH, 0, stream>>>(Uxb, aggn, aggd, xtmp, stats + 2 * NH, stats + 3 * NH);
    k_bn_apply_x<<<NB * NV, NH, 0, stream>>>(x, xtmp, stats + 2 * NH, stats + 3 * NH, F(p0 + 10), F(p0 + 11));
    k_bn_apply_e<<<2048, 256, 0, stream>>>(e, etmp, stats, stats + NH, F(p0 + 12), F(p0 + 13),
                                           (long)NB * NV * NV * NH);
  }

  // --- quad gather + MLP head (f16 activations) ---
  k_tour_edges<<<NB, 32, 0, stream>>>(tour, tiu, tiv);
  k_quad_gather<<<NB * NP, NH, 0, stream>>>(e, vals, xdir, tiu, tiv, F(54), F(55), feat, hb);
  const int gm = (int)((Mrows + 127) / 128);
  // acl layer 2 (linear) writes cost_emb into feat[:,512:640]
  gemm_n128<_Float16, _Float16, false><<<gm, 256, 0, stream>>>(hb, (int)Mrows, NH, Wacl1, F(57), NH, feat + 512, 640);
  // pre: 640 -> 128 (relu), 128 -> 128 (linear)
  gemm_n128<_Float16, _Float16, true ><<<gm, 256, 0, stream>>>(feat, (int)Mrows, 640, Wpre0, F(59), 640, s1, NH);
  gemm_n128<_Float16, _Float16, false><<<gm, 256, 0, stream>>>(s1, (int)Mrows, NH, Wpre1, F(61), NH, s2, NH);
  // act: 4x (128 -> 128, relu)
  gemm_n128<_Float16, _Float16, true ><<<gm, 256, 0, stream>>>(s2, (int)Mrows, NH, Wact[0], F(63), NH, s1, NH);
  gemm_n128<_Float16, _Float16, true ><<<gm, 256, 0, stream>>>(s1, (int)Mrows, NH, Wact[1], F(65), NH, s2, NH);
  gemm_n128<_Float16, _Float16, true ><<<gm, 256, 0, stream>>>(s2, (int)Mrows, NH, Wact[2], F(67), NH, s1, NH);
  gemm_n128<_Float16, _Float16, true ><<<gm, 256, 0, stream>>>(s1, (int)Mrows, NH, Wact[3], F(69), NH, s2, NH);
  // final 128 -> 1
  k_rowdot<<<(int)Mrows, NH, 0, stream>>>(s2, F(70), F(71), logits);
  // argmax + log-softmax + edge recovery -> d_out (edges 144 | pi 24 | actions 24)
  k_select<<<NB, 256, 0, stream>>>(logits, tiu, tiv, (float*)d_out);
}
